// Attention_77051713290847
// MI455X (gfx1250) — compile-verified, matrix operand order checked by reading, not run on hardware
//
#include <hip/hip_runtime.h>
#include <stdint.h>

typedef __attribute__((ext_vector_type(16))) __bf16 v16bf;
typedef __attribute__((ext_vector_type(8)))  float  v8f;

#define BQ 2
#define SQ 2048
#define DQ 4096
#define RQ 4
#define KQ 8
#define HQ 128
#define BS (BQ*SQ)     /* 4096 */
#define QN (RQ*KQ*HQ)  /* 4096 query-head feature dim */
#define KN (KQ*HQ)     /* 1024 kv feature dim */

__device__ __forceinline__ unsigned short f2bf(float f) {
  unsigned int u = __float_as_uint(f);
  u += 0x7FFFu + ((u >> 16) & 1u);          // round-to-nearest-even
  return (unsigned short)(u >> 16);
}
__device__ __forceinline__ float bf2f(unsigned short h) {
  return __uint_as_float(((unsigned int)h) << 16);
}

union FragBF { uint4 u[2]; v16bf v; };

__device__ __forceinline__ v8f wmma_bf16(v16bf a, v16bf b, v8f c) {
  return __builtin_amdgcn_wmma_f32_16x16x32_bf16(false, a, false, b, (short)0, c,
                                                 false, false);
}

// ---------------------------------------------------------------- converts
__global__ void f32_to_bf16_kernel(const float* __restrict__ in,
                                   unsigned short* __restrict__ out, long n) {
  long i = (long)blockIdx.x * blockDim.x + threadIdx.x;
  long stride = (long)gridDim.x * blockDim.x;
  for (; i < n; i += stride) out[i] = f2bf(in[i]);
}

// ---------------------------------------------------------------- RoPE (in place)
// t layout: [B*S, NH, H] bf16. Each thread handles one (h, h+H/2) pair.
__global__ void rope_kernel(unsigned short* __restrict__ t,
                            const float* __restrict__ sn,
                            const float* __restrict__ cs,
                            int NH, float scale) {
  long idx = (long)blockIdx.x * blockDim.x + threadIdx.x;
  long total = (long)BS * NH * (HQ / 2);
  if (idx >= total) return;
  int hp = (int)(idx % (HQ / 2));
  long r = idx / (HQ / 2);
  int head = (int)(r % NH);
  long row = r / NH;                 // b*S + s
  int s = (int)(row % SQ);
  unsigned short* p = t + (row * NH + head) * HQ;
  float v1 = bf2f(p[hp]);
  float v2 = bf2f(p[hp + HQ / 2]);
  float c1 = cs[s * HQ + hp],          s1 = sn[s * HQ + hp];
  float c2 = cs[s * HQ + hp + HQ / 2], s2 = sn[s * HQ + hp + HQ / 2];
  p[hp]          = f2bf((v1 * c1 - v2 * s1) * scale);
  p[hp + HQ / 2] = f2bf((v2 * c2 + v1 * s2) * scale);
}

// ---------------------------------------------------------------- V transpose
// vb: [B*S, K*H] -> vt: [B, K, H, S]
__global__ void transpose_v_kernel(const unsigned short* __restrict__ vb,
                                   unsigned short* __restrict__ vt) {
  long idx = (long)blockIdx.x * blockDim.x + threadIdx.x;
  long total = (long)BQ * KQ * HQ * SQ;
  if (idx >= total) return;
  int s = (int)(idx % SQ);
  long r = idx / SQ;
  int h = (int)(r % HQ); r /= HQ;
  int k = (int)(r % KQ);
  int b = (int)(r / KQ);
  vt[idx] = vb[((long)(b * SQ + s)) * KN + k * HQ + h];
}

// ---------------------------------------------------------------- bf16 GEMM
// C[M,N] = A[M,Kd] * B[Kd,N], A/B bf16 row-major, C f32 or bf16.
// Block tile 128x128, 8 waves of 32(M)x64(N), K-step 32.
template <bool OUT_F32>
__global__ __launch_bounds__(256) void gemm_bf16_kernel(
    const unsigned short* __restrict__ A, const unsigned short* __restrict__ Bm,
    void* __restrict__ Cv, int M, int N, int Kd) {
  __shared__ __align__(16) unsigned short As[128][40];
  __shared__ __align__(16) unsigned short BsT[128][40];
  const int tid = threadIdx.x;
  const int lane = tid & 31, w = tid >> 5;
  const int hf = lane >> 4, l16 = lane & 15;
  const int wm = w >> 1, wn = w & 1;
  const long m0 = (long)blockIdx.y * 128, n0 = (long)blockIdx.x * 128;
  v8f acc[2][4] = {};
  for (int k0 = 0; k0 < Kd; k0 += 32) {
    __syncthreads();
#pragma unroll
    for (int i = 0; i < 2; ++i) {           // stage A 128x32
      int u = tid + i * 256;
      int r = u >> 2, c = (u & 3) << 3;
      *(uint4*)&As[r][c] = *(const uint4*)&A[(m0 + r) * Kd + k0 + c];
    }
#pragma unroll
    for (int i = 0; i < 2; ++i) {           // stage B 32x128 transposed
      int u = tid + i * 256;
      int kk = u >> 4, n8 = (u & 15) << 3;
      uint4 d = *(const uint4*)&Bm[(long)(k0 + kk) * N + n0 + n8];
      const unsigned short* dsv = (const unsigned short*)&d;
#pragma unroll
      for (int j = 0; j < 8; ++j) BsT[n8 + j][kk] = dsv[j];
    }
    __syncthreads();
    v16bf af[2], bfr[4];
#pragma unroll
    for (int i = 0; i < 2; ++i) {
      FragBF f;
      const unsigned short* p = &As[wm * 32 + i * 16 + l16][0];
      f.u[0] = *(const uint4*)(p + hf * 8);
      f.u[1] = *(const uint4*)(p + 16 + hf * 8);
      af[i] = f.v;
    }
#pragma unroll
    for (int j = 0; j < 4; ++j) {
      FragBF f;
      const unsigned short* p = &BsT[wn * 64 + j * 16 + l16][0];
      f.u[0] = *(const uint4*)(p + hf * 8);
      f.u[1] = *(const uint4*)(p + 16 + hf * 8);
      bfr[j] = f.v;
    }
#pragma unroll
    for (int i = 0; i < 2; ++i)
#pragma unroll
      for (int j = 0; j < 4; ++j) acc[i][j] = wmma_bf16(af[i], bfr[j], acc[i][j]);
  }
#pragma unroll
  for (int i = 0; i < 2; ++i)
#pragma unroll
    for (int j = 0; j < 4; ++j)
#pragma unroll
      for (int v = 0; v < 8; ++v) {
        const long row = m0 + wm * 32 + i * 16 + v + hf * 8;
        const long col = n0 + wn * 64 + j * 16 + l16;
        if (OUT_F32)
          ((float*)Cv)[row * N + col] = acc[i][j][v];
        else
          ((unsigned short*)Cv)[row * N + col] = f2bf(acc[i][j][v]);
      }
}

// ---------------------------------------------------------------- flash attention
// Q:[B*S,QN] roped bf16, Kr:[B*S,K,H] roped+scaled bf16, Vt:[B,K,H,S] bf16,
// Y:[B*S,QN] bf16. Each wave owns 16 query rows; loops over 32-key causal tiles.
__global__ __launch_bounds__(256) void flash_attn_kernel(
    const unsigned short* __restrict__ Q, const unsigned short* __restrict__ Kr,
    const unsigned short* __restrict__ Vt, unsigned short* __restrict__ Y) {
  __shared__ __align__(16) unsigned short Pld[8][16][40];  // per-wave P tile
  const int tid = threadIdx.x;
  const int lane = tid & 31;
  const int w = tid >> 5;
  const int hf = lane >> 4;
  const int l16 = lane & 15;
  const int b = blockIdx.y / (RQ * KQ);
  const int rk = blockIdx.y % (RQ * KQ);
  const int kh = rk % KQ;
  const int qb0 = blockIdx.x * 128 + w * 16;

  // Q fragments: 16 rows x 128 h as 4 A-fragments (16x32)
  v16bf qf[4];
  {
    const unsigned short* qrow = Q + ((long)(b * SQ + qb0 + l16)) * QN + rk * HQ;
#pragma unroll
    for (int c = 0; c < 4; ++c) {
      FragBF f;
      f.u[0] = *(const uint4*)(qrow + c * 32 + hf * 8);
      f.u[1] = *(const uint4*)(qrow + c * 32 + 16 + hf * 8);
      qf[c] = f.v;
    }
  }

  v8f o[8] = {};
  float mrow[8], lrow[8];
#pragma unroll
  for (int v = 0; v < 8; ++v) { mrow[v] = -1e30f; lrow[v] = 0.0f; }

  const int tEnd = qb0 + 16;
  for (int t0 = 0; t0 < tEnd; t0 += 32) {
    v8f s0 = {}, s1 = {};
#pragma unroll
    for (int c = 0; c < 4; ++c) {  // scores: Q(16x128) x K^T(128x32)
      const unsigned short* k0p =
          Kr + (((long)(b * SQ + t0 + l16)) * KQ + kh) * HQ + c * 32;
      FragBF f;
      f.u[0] = *(const uint4*)(k0p + hf * 8);
      f.u[1] = *(const uint4*)(k0p + 16 + hf * 8);
      s0 = wmma_bf16(qf[c], f.v, s0);
      const unsigned short* k1p = k0p + (long)16 * KQ * HQ;
      f.u[0] = *(const uint4*)(k1p + hf * 8);
      f.u[1] = *(const uint4*)(k1p + 16 + hf * 8);
      s1 = wmma_bf16(qf[c], f.v, s1);
    }
    // online softmax (rows live on 16-lane half-groups, 8 rows per lane)
#pragma unroll
    for (int v = 0; v < 8; ++v) {
      const int qrow = qb0 + v + hf * 8;
      float a0 = (t0 + l16 <= qrow) ? s0[v] : -3.0e38f;
      float a1 = (t0 + 16 + l16 <= qrow) ? s1[v] : -3.0e38f;
      float mx = fmaxf(a0, a1);
#pragma unroll
      for (int off = 1; off < 16; off <<= 1) mx = fmaxf(mx, __shfl_xor(mx, off, 32));
      const float mnew = fmaxf(mrow[v], mx);
      const float corr = __expf(mrow[v] - mnew);
      const float p0 = __expf(a0 - mnew);
      const float p1 = __expf(a1 - mnew);
      float sum = p0 + p1;
#pragma unroll
      for (int off = 1; off < 16; off <<= 1) sum += __shfl_xor(sum, off, 32);
      lrow[v] = lrow[v] * corr + sum;
      mrow[v] = mnew;
      Pld[w][v + hf * 8][l16] = f2bf(p0);
      Pld[w][v + hf * 8][16 + l16] = f2bf(p1);
#pragma unroll
      for (int j = 0; j < 8; ++j) o[j][v] *= corr;
    }
    // wave-private LDS transpose: C-layout P -> A-layout fragment
    asm volatile("s_wait_dscnt 0x0" ::: "memory");
    FragBF pf;
    pf.u[0] = *(const uint4*)&Pld[w][l16][hf * 8];
    pf.u[1] = *(const uint4*)&Pld[w][l16][16 + hf * 8];
    // O += P(16x32) x V(32x128)
#pragma unroll
    for (int j = 0; j < 8; ++j) {
      const unsigned short* vp =
          Vt + (((long)(b * KQ + kh)) * HQ + j * 16 + l16) * SQ + t0;
      FragBF f;
      f.u[0] = *(const uint4*)(vp + hf * 8);
      f.u[1] = *(const uint4*)(vp + 16 + hf * 8);
      o[j] = wmma_bf16(pf.v, f.v, o[j]);
    }
  }
  // epilogue: normalize and store bf16
#pragma unroll
  for (int v = 0; v < 8; ++v) {
    const float inv = 1.0f / lrow[v];
#pragma unroll
    for (int j = 0; j < 8; ++j) o[j][v] *= inv;
  }
#pragma unroll
  for (int j = 0; j < 8; ++j)
#pragma unroll
    for (int v = 0; v < 8; ++v) {
      const int row = qb0 + v + hf * 8;
      const int col = rk * HQ + j * 16 + l16;
      Y[(long)(b * SQ + row) * QN + col] = f2bf(o[j][v]);
    }
}

// ---------------------------------------------------------------- launch
extern "C" void kernel_launch(void* const* d_in, const int* in_sizes, int n_in,
                              void* d_out, int out_size, void* d_ws, size_t ws_size,
                              hipStream_t stream) {
  (void)in_sizes; (void)n_in; (void)out_size; (void)ws_size;
  const float* x  = (const float*)d_in[0];
  const float* wq = (const float*)d_in[1];
  const float* wk = (const float*)d_in[2];
  const float* wv = (const float*)d_in[3];
  const float* wo = (const float*)d_in[4];
  /* d_in[5] = mask: causal handled analytically */
  const float* sinb = (const float*)d_in[6];
  const float* cosb = (const float*)d_in[7];

  char* ws = (char*)d_ws;
  auto carve = [&](size_t bytes) {
    char* p = ws;
    ws += (bytes + 255) & ~(size_t)255;
    return p;
  };
  unsigned short* xb  = (unsigned short*)carve((size_t)BS * DQ * 2);
  unsigned short* wqb = (unsigned short*)carve((size_t)DQ * QN * 2);
  unsigned short* wkb = (unsigned short*)carve((size_t)DQ * KN * 2);
  unsigned short* wvb = (unsigned short*)carve((size_t)DQ * KN * 2);
  unsigned short* wob = (unsigned short*)carve((size_t)QN * DQ * 2);
  unsigned short* qb  = (unsigned short*)carve((size_t)BS * QN * 2);
  unsigned short* kb  = (unsigned short*)carve((size_t)BS * KN * 2);
  unsigned short* vb  = (unsigned short*)carve((size_t)BS * KN * 2);
  unsigned short* vt  = (unsigned short*)carve((size_t)BQ * KQ * HQ * SQ * 2);
  unsigned short* yb  = (unsigned short*)carve((size_t)BS * QN * 2);

  // 1) f32 -> bf16
  f32_to_bf16_kernel<<<1024, 256, 0, stream>>>(x,  xb,  (long)BS * DQ);
  f32_to_bf16_kernel<<<1024, 256, 0, stream>>>(wq, wqb, (long)DQ * QN);
  f32_to_bf16_kernel<<<1024, 256, 0, stream>>>(wk, wkb, (long)DQ * KN);
  f32_to_bf16_kernel<<<1024, 256, 0, stream>>>(wv, wvb, (long)DQ * KN);
  f32_to_bf16_kernel<<<1024, 256, 0, stream>>>(wo, wob, (long)QN * DQ);

  // 2) projections (WMMA GEMMs)
  gemm_bf16_kernel<false><<<dim3(QN / 128, BS / 128), 256, 0, stream>>>(
      xb, wqb, qb, BS, QN, DQ);
  gemm_bf16_kernel<false><<<dim3(KN / 128, BS / 128), 256, 0, stream>>>(
      xb, wkb, kb, BS, KN, DQ);
  gemm_bf16_kernel<false><<<dim3(KN / 128, BS / 128), 256, 0, stream>>>(
      xb, wvb, vb, BS, KN, DQ);

  // 3) RoPE (scale H^-0.5 folded into K)
  {
    long nq = (long)BS * (RQ * KQ) * (HQ / 2);
    rope_kernel<<<(unsigned)((nq + 255) / 256), 256, 0, stream>>>(
        qb, sinb, cosb, RQ * KQ, 1.0f);
    long nk = (long)BS * KQ * (HQ / 2);
    rope_kernel<<<(unsigned)((nk + 255) / 256), 256, 0, stream>>>(
        kb, sinb, cosb, KQ, 0.08838834764831845f);
  }

  // 4) V transpose for contiguous PV fragments
  {
    long nv = (long)BQ * KQ * HQ * SQ;
    transpose_v_kernel<<<(unsigned)((nv + 255) / 256), 256, 0, stream>>>(vb, vt);
  }

  // 5) flash attention
  flash_attn_kernel<<<dim3(SQ / 128, BQ * RQ * KQ), 256, 0, stream>>>(qb, kb, vt, yb);

  // 6) output projection -> f32 d_out
  gemm_bf16_kernel<true><<<dim3(DQ / 128, BS / 128), 256, 0, stream>>>(
      yb, wob, (float*)d_out, BS, DQ, QN);
}
